// WindowMultiHeadedAttention_62371515073137
// MI455X (gfx1250) — matrix operand order repaired and checked by reading, not compiled, not run on hardware
//
#include <hip/hip_runtime.h>
#include <hip/hip_bf16.h>

// ---------------------------------------------------------------------------
// Window MHA for MI455X (gfx1250, wave32, WMMA bf16)
//   H=W=32, ZONE=2, HEAD=8, C=512, BT=8, T=4 -> b=2, windows=8, L=1024, c_h=64
// ---------------------------------------------------------------------------

typedef __bf16 bf16_t;
typedef __attribute__((ext_vector_type(8)))  __bf16 v8bf;
typedef __attribute__((ext_vector_type(16))) __bf16 v16bf;
typedef __attribute__((ext_vector_type(8)))  float  v8f;

#define CDIM   512
#define NROWS  8192      // BT * N = 8 * 1024
#define LWIN   1024      // T * 16 * 16

// ---- WMMA fragment helpers (layouts per CDNA5 ISA §7.12.2) ------

// A (16x32 bf16): lane holds row lane%16; elems e0..7 at K = 8*half + e,
// e8..15 at K = 16 + 8*half + e  (half = lane>>4). Two b128 loads.
__device__ __forceinline__ v16bf load_a_frag(const bf16_t* __restrict__ rowp, int half) {
    union { v16bf v; v8bf h[2]; } u;
    u.h[0] = *(const v8bf*)(rowp + half * 8);
    u.h[1] = *(const v8bf*)(rowp + 16 + half * 8);
    return u.v;
}

// B (32x16 bf16): lane holds column lane%16; elems e0..7 at K = 8*half + e, e8..15 at K = 16 + 8*half + e.
// Row-major W row == B column for X @ W^T (no transpose anywhere).
__device__ __forceinline__ v16bf load_b_frag(const bf16_t* __restrict__ colp, int half) {
    union { v16bf v; v8bf h[2]; } u;
    u.h[0] = *(const v8bf*)(colp + half * 8);
    u.h[1] = *(const v8bf*)(colp + 16 + half * 8);
    return u.v;
}

__device__ __forceinline__ v8f wmma_bf16(v16bf a, v16bf b, v8f c) {
    // (neg_a, A, neg_b, B, c_mod, C, reuse_a, reuse_b)
    return __builtin_amdgcn_wmma_f32_16x16x32_bf16(false, a, false, b,
                                                   (short)0, c, false, false);
}

// ---- fp32 -> bf16 bulk convert (8 elems/thread, b128 in / b128 out) --------
__global__ __launch_bounds__(256) void cvt_kernel(const float* __restrict__ in,
                                                  bf16_t* __restrict__ out, int n) {
    int i = (blockIdx.x * 256 + threadIdx.x) * 8;
    if (i + 8 <= n) {
        float4 f0 = *(const float4*)(in + i);
        float4 f1 = *(const float4*)(in + i + 4);
        v8bf o;
        o[0] = (bf16_t)f0.x; o[1] = (bf16_t)f0.y; o[2] = (bf16_t)f0.z; o[3] = (bf16_t)f0.w;
        o[4] = (bf16_t)f1.x; o[5] = (bf16_t)f1.y; o[6] = (bf16_t)f1.z; o[7] = (bf16_t)f1.w;
        *(v8bf*)(out + i) = o;
    }
}

// ---- Fused QKV projection, windowed outputs --------------------------------
// Wave tile: 64 rows x 16 cols x {Q,K,V}: 4 A-frags shared by 3 B-frags
// -> 14 b128 loads per 12 WMMAs.  q_w,k_w: [win][m][c]; vt_w: [win][c][m].
__global__ __launch_bounds__(256) void qkv_kernel(
    const bf16_t* __restrict__ xb,
    const bf16_t* __restrict__ wq, const bf16_t* __restrict__ wk, const bf16_t* __restrict__ wv,
    const float*  __restrict__ bq, const float*  __restrict__ bk, const float*  __restrict__ bv,
    bf16_t* __restrict__ q_w, bf16_t* __restrict__ k_w, bf16_t* __restrict__ vt_w) {
    const int lane = threadIdx.x & 31;
    const int wave = threadIdx.x >> 5;
    const int tile = blockIdx.x * 8 + wave;     // 4096 = 128 rowblocks x 32 coltiles
    const int rb = tile >> 5, ct = tile & 31;
    const int rowbase = rb * 64, jbase = ct * 16;
    const int ln = lane & 15, half = lane >> 4;

    const bf16_t* arow0 = xb + (size_t)(rowbase + ln) * CDIM;
    const bf16_t* bqrow = wq + (size_t)(jbase + ln) * CDIM;
    const bf16_t* bkrow = wk + (size_t)(jbase + ln) * CDIM;
    const bf16_t* bvrow = wv + (size_t)(jbase + ln) * CDIM;

    v8f accq[4] = {}; v8f acck[4] = {}; v8f accv[4] = {};
    for (int kb = 0; kb < CDIM; kb += 32) {
        v16bf b0 = load_b_frag(bqrow + kb, half);
        v16bf b1 = load_b_frag(bkrow + kb, half);
        v16bf b2 = load_b_frag(bvrow + kb, half);
        __builtin_prefetch(arow0 + kb + 32, 0, 1);   // global_prefetch_b8
#pragma unroll
        for (int s = 0; s < 4; ++s) {
            v16bf a = load_a_frag(arow0 + (size_t)s * 16 * CDIM + kb, half);
            accq[s] = wmma_bf16(a, b0, accq[s]);
            acck[s] = wmma_bf16(a, b1, acck[s]);
            accv[s] = wmma_bf16(a, b2, accv[s]);
        }
    }

    const float biq = bq[jbase + ln], bik = bk[jbase + ln], biv = bv[jbase + ln];
    const int Mb = half * 8;                    // D: lane holds M = v + 8*half, N = ln

#pragma unroll
    for (int s = 0; s < 4; ++s) {
        // Window mapping: uniform inside each 16-row subtile.
        int rbr = rowbase + s * 16;
        int bt = rbr >> 10, n0 = rbr & 1023;
        int y = n0 >> 5, x0 = n0 & 31;          // x0 in {0,16}
        int b_i = bt >> 2, t_i = bt & 3;
        int win = b_i * 4 + (y >> 4) * 2 + (x0 >> 4);
        int m0  = t_i * 256 + (y & 15) * 16;

        bf16_t* qp = q_w + ((size_t)win * LWIN + m0 + Mb) * CDIM + jbase + ln;
        bf16_t* kp = k_w + ((size_t)win * LWIN + m0 + Mb) * CDIM + jbase + ln;
#pragma unroll
        for (int v = 0; v < 8; ++v) {
            qp[v * CDIM] = (bf16_t)(accq[s][v] + biq);
            kp[v * CDIM] = (bf16_t)(acck[s][v] + bik);
        }
        v8bf pv;                                // 8 consecutive m -> one b128 store
#pragma unroll
        for (int v = 0; v < 8; ++v) pv[v] = (bf16_t)(accv[s][v] + biv);
        *(v8bf*)(vt_w + ((size_t)win * CDIM + jbase + ln) * LWIN + m0 + Mb) = pv;
    }
}

// ---- Attention: one block per (win, head, 32-row q block) ------------------
// S = QK^T in LDS (32x1024 f32 = 128 KB), softmax in place, then O = P @ V.
__global__ __launch_bounds__(256) void attn_kernel(
    const bf16_t* __restrict__ q_w, const bf16_t* __restrict__ k_w,
    const bf16_t* __restrict__ vt_w, bf16_t* __restrict__ att_w) {
    extern __shared__ float S[];                // 32 * 1024 floats
    const int lane = threadIdx.x & 31;
    const int wave = threadIdx.x >> 5;
    const int ln = lane & 15, half = lane >> 4;

    const int wh  = blockIdx.x >> 5;            // 0..63
    const int rb  = blockIdx.x & 31;            // 0..31
    const int win = wh >> 3, head = wh & 7;
    const int l0  = rb * 32;

    const bf16_t* qbase = q_w + (size_t)win * LWIN * CDIM + head * 64;
    const bf16_t* kbase = k_w + (size_t)win * LWIN * CDIM + head * 64;

    // Phase 1: scores. Wave keeps its 2 Q A-frags (K=c_h=64) in registers and
    // sweeps 16 column tiles: 4 + 64 loads per 32 WMMAs.
    {
        int rt = wave & 1;
        int ctbase = (wave >> 1) * 16;
        const bf16_t* arow = qbase + (size_t)(l0 + rt * 16 + ln) * CDIM;
        v16bf a0 = load_a_frag(arow, half);
        v16bf a1 = load_a_frag(arow + 32, half);
#pragma unroll 4
        for (int c = 0; c < 16; ++c) {
            int ctn = ctbase + c;
            const bf16_t* brow = kbase + (size_t)(ctn * 16 + ln) * CDIM;
            v8f acc = {};
            acc = wmma_bf16(a0, load_b_frag(brow, half),      acc);
            acc = wmma_bf16(a1, load_b_frag(brow + 32, half), acc);
            float* sp = S + (size_t)(rt * 16 + half * 8) * 1024 + ctn * 16 + ln;
#pragma unroll
            for (int v = 0; v < 8; ++v) sp[v * 1024] = acc[v];
        }
    }
    __syncthreads();

    // Phase 2: softmax( S / 8 ) rows; 8 threads/row, wave32 shfl reductions.
    {
        int row = threadIdx.x >> 3, sub = threadIdx.x & 7;
        float* rp = S + (size_t)row * 1024;
        float mx = -3.0e38f;
        for (int m = sub; m < 1024; m += 8) mx = fmaxf(mx, rp[m]);
        mx = fmaxf(mx, __shfl_xor(mx, 1, 32));
        mx = fmaxf(mx, __shfl_xor(mx, 2, 32));
        mx = fmaxf(mx, __shfl_xor(mx, 4, 32));
        float sum = 0.f;
        for (int m = sub; m < 1024; m += 8) {
            float e = __expf(0.125f * (rp[m] - mx));
            rp[m] = e; sum += e;
        }
        sum += __shfl_xor(sum, 1, 32);
        sum += __shfl_xor(sum, 2, 32);
        sum += __shfl_xor(sum, 4, 32);
        float inv = 1.0f / sum;
        for (int m = sub; m < 1024; m += 8) rp[m] *= inv;
    }
    __syncthreads();

    // Phase 3: O = P @ V. 8 tiles (2 rt x 4 c-tiles), one per wave; K = 1024.
    {
        int rt = wave >> 2, ctc = wave & 3;
        const bf16_t* vrow = vt_w + ((size_t)win * CDIM + head * 64 + ctc * 16 + ln) * LWIN;
        v8f acc = {};
        for (int kb = 0; kb < 1024; kb += 32) {
            const float* prow = S + (size_t)(rt * 16 + ln) * 1024 + kb;
            v16bf a;
#pragma unroll
            for (int e = 0; e < 8; ++e) {
                a[e]     = (bf16_t)prow[half * 8 + e];
                a[8 + e] = (bf16_t)prow[16 + half * 8 + e];
            }
            v16bf b = load_b_frag(vrow + kb, half);  // vt row: contiguous in m
            acc = wmma_bf16(a, b, acc);
        }
        bf16_t* op = att_w + ((size_t)win * LWIN + l0 + rt * 16 + half * 8) * CDIM
                           + head * 64 + ctc * 16 + ln;
#pragma unroll
        for (int v = 0; v < 8; ++v) op[v * CDIM] = (bf16_t)acc[v];
    }
}

// ---- Output projection: out = att @ Wo^T + bo, un-windowing fused ----------
// Wave tile: 64 rows x 32 cols: 4 A-frags x 2 B-frags -> 12 loads / 8 WMMAs.
__global__ __launch_bounds__(256) void oproj_kernel(
    const bf16_t* __restrict__ att_w, const bf16_t* __restrict__ wo,
    const float* __restrict__ bo, float* __restrict__ out) {
    const int lane = threadIdx.x & 31;
    const int wave = threadIdx.x >> 5;
    const int tile = blockIdx.x * 8 + wave;     // 2048 = 128 rowblocks x 16 colpairs
    const int rb = tile >> 4, cp = tile & 15;
    const int rowbase = rb * 64, j0 = cp * 32;
    const int ln = lane & 15, half = lane >> 4;

    // Per-lane inverse window mapping for the 4 A-row subtiles.
    const bf16_t* arow[4];
#pragma unroll
    for (int s = 0; s < 4; ++s) {
        int r = rowbase + s * 16 + ln;
        int bt = r >> 10, n0 = r & 1023;
        int y = n0 >> 5, x = n0 & 31;
        int b_i = bt >> 2, t_i = bt & 3;
        int win = b_i * 4 + (y >> 4) * 2 + (x >> 4);
        int m   = t_i * 256 + (y & 15) * 16 + (x & 15);
        arow[s] = att_w + ((size_t)win * LWIN + m) * CDIM;
    }
    const bf16_t* brow0 = wo + (size_t)(j0 + ln) * CDIM;
    const bf16_t* brow1 = wo + (size_t)(j0 + 16 + ln) * CDIM;

    v8f acc[4][2] = {};
    for (int kb = 0; kb < CDIM; kb += 32) {
        v16bf bb0 = load_b_frag(brow0 + kb, half);
        v16bf bb1 = load_b_frag(brow1 + kb, half);
        __builtin_prefetch(brow0 + kb + 32, 0, 1);
#pragma unroll
        for (int s = 0; s < 4; ++s) {
            v16bf a = load_a_frag(arow[s] + kb, half);
            acc[s][0] = wmma_bf16(a, bb0, acc[s][0]);
            acc[s][1] = wmma_bf16(a, bb1, acc[s][1]);
        }
    }
    const float bj0 = bo[j0 + ln], bj1 = bo[j0 + 16 + ln];
#pragma unroll
    for (int s = 0; s < 4; ++s) {
        float* op = out + (size_t)(rowbase + s * 16 + half * 8) * CDIM;
#pragma unroll
        for (int v = 0; v < 8; ++v) {
            op[v * CDIM + j0 + ln]      = acc[s][0][v] + bj0;
            op[v * CDIM + j0 + 16 + ln] = acc[s][1][v] + bj1;
        }
    }
}

// ---------------------------------------------------------------------------
extern "C" void kernel_launch(void* const* d_in, const int* in_sizes, int n_in,
                              void* d_out, int out_size, void* d_ws, size_t ws_size,
                              hipStream_t stream) {
    (void)in_sizes; (void)n_in; (void)out_size; (void)ws_size;
    const float* x  = (const float*)d_in[0];
    // d_in[1] = t (hardcoded T=4)
    const float* Wq = (const float*)d_in[2];
    const float* bq = (const float*)d_in[3];
    const float* Wk = (const float*)d_in[4];
    const float* bk = (const float*)d_in[5];
    const float* Wv = (const float*)d_in[6];
    const float* bv = (const float*)d_in[7];
    const float* Wo = (const float*)d_in[8];
    const float* bo = (const float*)d_in[9];
    float* out = (float*)d_out;

    const size_t XE = (size_t)NROWS * CDIM;      // 4,194,304 elems
    const size_t WE = (size_t)CDIM * CDIM;       //   262,144 elems
    char* ws = (char*)d_ws;
    bf16_t* xb   = (bf16_t*)(ws);
    bf16_t* wqb  = (bf16_t*)(ws + 2 * XE);
    bf16_t* wkb  = (bf16_t*)(ws + 2 * (XE + WE));
    bf16_t* wvb  = (bf16_t*)(ws + 2 * (XE + 2 * WE));
    bf16_t* wob  = (bf16_t*)(ws + 2 * (XE + 3 * WE));
    bf16_t* q_w  = (bf16_t*)(ws + 2 * (XE + 4 * WE));
    bf16_t* k_w  = (bf16_t*)(ws + 2 * (2 * XE + 4 * WE));
    bf16_t* vt_w = (bf16_t*)(ws + 2 * (3 * XE + 4 * WE));
    bf16_t* attw = (bf16_t*)(ws + 2 * (4 * XE + 4 * WE));

    cvt_kernel<<<(int)(XE / (8 * 256)), 256, 0, stream>>>(x,  xb,  (int)XE);
    cvt_kernel<<<(int)(WE / (8 * 256)), 256, 0, stream>>>(Wq, wqb, (int)WE);
    cvt_kernel<<<(int)(WE / (8 * 256)), 256, 0, stream>>>(Wk, wkb, (int)WE);
    cvt_kernel<<<(int)(WE / (8 * 256)), 256, 0, stream>>>(Wv, wvb, (int)WE);
    cvt_kernel<<<(int)(WE / (8 * 256)), 256, 0, stream>>>(Wo, wob, (int)WE);

    qkv_kernel <<<512,  256, 0, stream>>>(xb, wqb, wkb, wvb, bq, bk, bv, q_w, k_w, vt_w);
    attn_kernel<<<2048, 256, 32 * 1024 * sizeof(float), stream>>>(q_w, k_w, vt_w, attw);
    oproj_kernel<<<256, 256, 0, stream>>>(attw, wob, bo, out);
}
